// TransformerDecoderModel_15204184228682
// MI455X (gfx1250) — matrix operand order repaired and checked by reading, not compiled, not run on hardware
//
#include <hip/hip_runtime.h>
#include <hip/hip_bf16.h>

typedef __attribute__((ext_vector_type(16))) _Float16 v16h;
typedef __attribute__((ext_vector_type(8)))  float    v8f;

#define VOCAB  32000
#define SEQ    512
#define DMODEL 512
#define NHEAD  8
#define FFDIM  2048
#define NLAYER 2
#define NCOMP  4
#define DHEAD  64
#define NEGINF (-1.0e30f)

// ---------------------------------------------------------------------------
// WMMA fragment builders (f32 memory -> f16 fragments), contiguous k order (not the ISA table order), same for A and B:
//  A 16x32 f16 : lane l -> row M = l&15, half h = l>>4;
//                element j: K = 16h+j
//  B 32x16 f16 : lane l -> col N = l&15, half h = l>>4; element j: K = 16h+j
//  C/D 16x16 f32: element r -> (M = r + 8h, N = l&15)
// ---------------------------------------------------------------------------
__device__ __forceinline__ v16h load_a_frag(const float* __restrict__ row, int k0, int half) {
  v16h a;
  const float* p0 = row + k0 + 16 * half;
  const float* p1 = row + k0 + 16 * half + 8;
#pragma unroll
  for (int j = 0; j < 8; ++j) a[j] = (_Float16)p0[j];
#pragma unroll
  for (int j = 0; j < 8; ++j) a[8 + j] = (_Float16)p1[j];
  return a;
}

__device__ __forceinline__ v16h load_b_frag(const float* __restrict__ row, int k0, int half) {
  v16h b;
  const float* p = row + k0 + 16 * half;
#pragma unroll
  for (int j = 0; j < 16; ++j) b[j] = (_Float16)p[j];
  return b;
}

__device__ __forceinline__ v16h load_b_frag_strided(const float* __restrict__ base, int stride,
                                                    int k0, int half) {
  v16h b;
  const float* p = base + (size_t)(k0 + 16 * half) * stride;
#pragma unroll
  for (int j = 0; j < 16; ++j) b[j] = (_Float16)p[(size_t)j * stride];
  return b;
}

__device__ __forceinline__ float wave_sum(float v) {
#pragma unroll
  for (int off = 16; off > 0; off >>= 1) v += __shfl_xor(v, off, 32);
  return v;
}
__device__ __forceinline__ float wave_max(float v) {
#pragma unroll
  for (int off = 16; off > 0; off >>= 1) v = fmaxf(v, __shfl_xor(v, off, 32));
  return v;
}

// ---------------------------------------------------------------------------
// Fused custom-neuron: out[s,o] = comp[n*,o] where n* = argmax_n logit[n,o].
// One wave per 16x16 (s,o) tile; 4 logit + 4 comp WMMA accumulator chains.
// All 8 B fragments are materialized into distinct register sets BEFORE the
// WMMA burst so the 4-coexec-NOP WMMA->VALU WAR hazard (ISA 7.12.1) is paid
// at most once per k-chunk instead of once per WMMA.
// Weights are streamed exactly once; selection is lane-local in the D layout.
// ---------------------------------------------------------------------------
__global__ __launch_bounds__(32) void custom_neuron_kernel(
    const float* __restrict__ x,   // [S, I]
    const float* __restrict__ sW,  // [O*NCOMP, I]
    const float* __restrict__ sb,  // [O*NCOMP]
    const float* __restrict__ cW,  // [NCOMP, O, I]
    const float* __restrict__ cb,  // [NCOMP, O]
    float* __restrict__ out,       // [S, O]
    int O, int I, int do_relu)
{
  const int lane = threadIdx.x;
  const int col  = lane & 15;
  const int half = lane >> 4;
  const int o0   = blockIdx.x * 16;
  const int s0   = blockIdx.y * 16;

  v8f accL[NCOMP] = {};
  v8f accC[NCOMP] = {};

  const float* xrow = x + (size_t)(s0 + col) * I;
  const float* sw_row[NCOMP];
  const float* cw_row[NCOMP];
#pragma unroll
  for (int n = 0; n < NCOMP; ++n) {
    sw_row[n] = sW + ((size_t)(o0 + col) * NCOMP + n) * I;
    cw_row[n] = cW + ((size_t)n * O + (o0 + col)) * I;
  }

  for (int k0 = 0; k0 < I; k0 += 32) {
    // Prefetch next k-chunk of the streamed weight rows (global_prefetch_b8).
#pragma unroll
    for (int n = 0; n < NCOMP; ++n) {
      __builtin_prefetch(sw_row[n] + k0 + 16 * half + 32);
      __builtin_prefetch(cw_row[n] + k0 + 16 * half + 32);
    }

    // Materialize A + all 8 B fragments first (disjoint registers) ...
    v16h a = load_a_frag(xrow, k0, half);
    v16h bL[NCOMP], bC[NCOMP];
#pragma unroll
    for (int n = 0; n < NCOMP; ++n) bL[n] = load_b_frag(sw_row[n], k0, half);
#pragma unroll
    for (int n = 0; n < NCOMP; ++n) bC[n] = load_b_frag(cw_row[n], k0, half);

    // ... then fire the 8 WMMAs back-to-back (no intervening VALU writes to
    // their sources -> no per-WMMA hazard NOPs).
#pragma unroll
    for (int n = 0; n < NCOMP; ++n)
      accL[n] = __builtin_amdgcn_wmma_f32_16x16x32_f16(false, a, false, bL[n], (short)0,
                                                       accL[n], false, false);
#pragma unroll
    for (int n = 0; n < NCOMP; ++n)
      accC[n] = __builtin_amdgcn_wmma_f32_16x16x32_f16(false, a, false, bC[n], (short)0,
                                                       accC[n], false, false);
  }

  float sbv[NCOMP], cbv[NCOMP];
#pragma unroll
  for (int n = 0; n < NCOMP; ++n) {
    sbv[n] = sb[(size_t)(o0 + col) * NCOMP + n];
    cbv[n] = cb[(size_t)n * O + (o0 + col)];
  }

#pragma unroll
  for (int r = 0; r < 8; ++r) {
    float bestL = accL[0][r] + sbv[0];
    float bestC = accC[0][r] + cbv[0];
#pragma unroll
    for (int n = 1; n < NCOMP; ++n) {
      float lg = accL[n][r] + sbv[n];
      float cv = accC[n][r] + cbv[n];
      bool take = lg > bestL;                 // first-max wins, matches jnp.argmax
      bestL = take ? lg : bestL;
      bestC = take ? cv : bestC;
    }
    float val = do_relu ? fmaxf(bestC, 0.0f) : bestC;
    out[(size_t)(s0 + r + 8 * half) * O + (o0 + col)] = val;
  }
}

// ---------------------------------------------------------------------------
// Attention scores: sc[h,i,j] = (q[i,h,:]·k[j,h,:]) / 8, causal mask j>i -> -1e30
// ---------------------------------------------------------------------------
__global__ __launch_bounds__(32) void attn_scores_kernel(
    const float* __restrict__ q, const float* __restrict__ k, float* __restrict__ sc)
{
  const int lane = threadIdx.x, col = lane & 15, half = lane >> 4;
  const int j0 = blockIdx.x * 16;
  const int s0 = blockIdx.y * 16;
  const int h  = blockIdx.z;

  v8f acc = {};
  const float* qrow = q + (size_t)(s0 + col) * DMODEL + h * DHEAD;
  const float* krow = k + (size_t)(j0 + col) * DMODEL + h * DHEAD;

  v16h a0 = load_a_frag(qrow, 0, half);
  v16h b0 = load_b_frag(krow, 0, half);
  v16h a1 = load_a_frag(qrow, 32, half);
  v16h b1 = load_b_frag(krow, 32, half);
  acc = __builtin_amdgcn_wmma_f32_16x16x32_f16(false, a0, false, b0, (short)0, acc, false, false);
  acc = __builtin_amdgcn_wmma_f32_16x16x32_f16(false, a1, false, b1, (short)0, acc, false, false);

#pragma unroll
  for (int r = 0; r < 8; ++r) {
    int i = s0 + r + 8 * half;
    int j = j0 + col;
    float v = (j > i) ? NEGINF : acc[r] * 0.125f;   // 1/sqrt(64)
    sc[((size_t)h * SEQ + i) * SEQ + j] = v;
  }
}

// ---------------------------------------------------------------------------
// Row softmax over SEQ columns (rows = NHEAD*SEQ), in place.
// ---------------------------------------------------------------------------
__global__ __launch_bounds__(256) void softmax_kernel(float* __restrict__ sc) {
  const int row = blockIdx.x;
  float* p = sc + (size_t)row * SEQ;
  __shared__ float red[8];
  __shared__ float s_val;
  const int tid = threadIdx.x;
  const int wid = tid >> 5;

  float m = -3.0e38f;
  for (int c = tid; c < SEQ; c += 256) m = fmaxf(m, p[c]);
  m = wave_max(m);
  if ((tid & 31) == 0) red[wid] = m;
  __syncthreads();
  if (tid == 0) {
    float mm = red[0];
#pragma unroll
    for (int i = 1; i < 8; ++i) mm = fmaxf(mm, red[i]);
    s_val = mm;
  }
  __syncthreads();
  const float mx = s_val;

  float part = 0.0f;
  for (int c = tid; c < SEQ; c += 256) {
    float e = expf(p[c] - mx);
    p[c] = e;
    part += e;
  }
  part = wave_sum(part);
  if ((tid & 31) == 0) red[wid] = part;
  __syncthreads();
  if (tid == 0) {
    float ss = 0.0f;
#pragma unroll
    for (int i = 0; i < 8; ++i) ss += red[i];
    s_val = ss;
  }
  __syncthreads();
  const float inv = 1.0f / s_val;
  for (int c = tid; c < SEQ; c += 256) p[c] *= inv;
}

// ---------------------------------------------------------------------------
// Context: ctx[s, h*64+d] = sum_j w[h,s,j] * v[j, h*64+d]   (WMMA over j)
// ---------------------------------------------------------------------------
__global__ __launch_bounds__(32) void attn_ctx_kernel(
    const float* __restrict__ w, const float* __restrict__ v, float* __restrict__ ctx)
{
  const int lane = threadIdx.x, col = lane & 15, half = lane >> 4;
  const int d0 = blockIdx.x * 16;
  const int s0 = blockIdx.y * 16;
  const int h  = blockIdx.z;

  v8f acc = {};
  const float* wrow  = w + ((size_t)h * SEQ + (s0 + col)) * SEQ;
  const float* vbase = v + (size_t)h * DHEAD + d0 + col;
  for (int k0 = 0; k0 < SEQ; k0 += 32) {
    v16h a = load_a_frag(wrow, k0, half);
    v16h b = load_b_frag_strided(vbase, DMODEL, k0, half);
    acc = __builtin_amdgcn_wmma_f32_16x16x32_f16(false, a, false, b, (short)0, acc, false, false);
  }
#pragma unroll
  for (int r = 0; r < 8; ++r) {
    ctx[(size_t)(s0 + r + 8 * half) * DMODEL + h * DHEAD + d0 + col] = acc[r];
  }
}

// ---------------------------------------------------------------------------
// Residual + LayerNorm, in place on x.
// ---------------------------------------------------------------------------
__global__ __launch_bounds__(256) void add_ln_kernel(
    float* __restrict__ x, const float* __restrict__ a,
    const float* __restrict__ g, const float* __restrict__ b)
{
  const int s = blockIdx.x;
  const int tid = threadIdx.x;
  const int wid = tid >> 5;
  __shared__ float buf[DMODEL];
  __shared__ float redA[8], redB[8];
  __shared__ float s_mu, s_rstd;

  float p1 = 0.0f, p2 = 0.0f;
  for (int c = tid; c < DMODEL; c += 256) {
    float y = x[(size_t)s * DMODEL + c] + a[(size_t)s * DMODEL + c];
    buf[c] = y;
    p1 += y;
    p2 += y * y;
  }
  p1 = wave_sum(p1);
  p2 = wave_sum(p2);
  if ((tid & 31) == 0) { redA[wid] = p1; redB[wid] = p2; }
  __syncthreads();
  if (tid == 0) {
    float s1 = 0.0f, s2 = 0.0f;
#pragma unroll
    for (int i = 0; i < 8; ++i) { s1 += redA[i]; s2 += redB[i]; }
    float mu  = s1 / (float)DMODEL;
    float var = s2 / (float)DMODEL - mu * mu;
    s_mu = mu;
    s_rstd = rsqrtf(var + 1e-5f);
  }
  __syncthreads();
  const float mu = s_mu, rstd = s_rstd;
  for (int c = tid; c < DMODEL; c += 256) {
    x[(size_t)s * DMODEL + c] = g[c] * (buf[c] - mu) * rstd + b[c];
  }
}

// ---------------------------------------------------------------------------
// Embedding * sqrt(D) + sinusoidal positional encoding.
// ---------------------------------------------------------------------------
__global__ __launch_bounds__(256) void embed_kernel(
    const int* __restrict__ ids, const float* __restrict__ emb, float* __restrict__ x)
{
  int idx = blockIdx.x * 256 + threadIdx.x;
  int s = idx / DMODEL;
  int d = idx % DMODEL;
  int tok = ids[s];
  int i2 = (d >> 1) << 1;   // 2*(d/2)
  float ang = (float)s * expf((float)i2 * (-9.210340371976184f / (float)DMODEL));
  float pe = (d & 1) ? cosf(ang) : sinf(ang);
  x[idx] = emb[(size_t)tok * DMODEL + d] * 22.62741699796952f + pe;  // sqrt(512)
}

// ---------------------------------------------------------------------------
// Orchestration
// ---------------------------------------------------------------------------
extern "C" void kernel_launch(void* const* d_in, const int* in_sizes, int n_in,
                              void* d_out, int out_size, void* d_ws, size_t ws_size,
                              hipStream_t stream) {
  (void)in_sizes; (void)n_in; (void)out_size; (void)ws_size;

  const int*   ids     = (const int*)  d_in[0];
  const float* emb     = (const float*)d_in[1];
  const float* qkvo_sW = (const float*)d_in[2];
  const float* qkvo_sb = (const float*)d_in[3];
  const float* qkvo_cW = (const float*)d_in[4];
  const float* qkvo_cb = (const float*)d_in[5];
  const float* l1_sW   = (const float*)d_in[6];
  const float* l1_sb   = (const float*)d_in[7];
  const float* l1_cW   = (const float*)d_in[8];
  const float* l1_cb   = (const float*)d_in[9];
  const float* l2_sW   = (const float*)d_in[10];
  const float* l2_sb   = (const float*)d_in[11];
  const float* l2_cW   = (const float*)d_in[12];
  const float* l2_cb   = (const float*)d_in[13];
  const float* ln1_g   = (const float*)d_in[14];
  const float* ln1_b   = (const float*)d_in[15];
  const float* ln2_g   = (const float*)d_in[16];
  const float* ln2_b   = (const float*)d_in[17];
  const float* out_sW  = (const float*)d_in[18];
  const float* out_sb  = (const float*)d_in[19];
  const float* out_cW  = (const float*)d_in[20];
  const float* out_cb  = (const float*)d_in[21];

  float* ws = (float*)d_ws;
  float* x  = ws;                                  // [S, D]
  float* qb = x  + (size_t)SEQ * DMODEL;           // [S, D]
  float* kb = qb + (size_t)SEQ * DMODEL;           // [S, D]
  float* vb = kb + (size_t)SEQ * DMODEL;           // [S, D]
  float* cx = vb + (size_t)SEQ * DMODEL;           // [S, D] attention context
  float* sc = cx + (size_t)SEQ * DMODEL;           // [H, S, S]
  float* hb = sc + (size_t)NHEAD * SEQ * SEQ;      // [S, FF]
  float* fb = hb + (size_t)SEQ * FFDIM;            // [S, D] residual branch

  embed_kernel<<<(SEQ * DMODEL) / 256, 256, 0, stream>>>(ids, emb, x);

  const size_t swsz = (size_t)DMODEL * NCOMP * DMODEL;
  const size_t sbsz = (size_t)DMODEL * NCOMP;
  const size_t cwsz = (size_t)NCOMP * DMODEL * DMODEL;
  const size_t cbsz = (size_t)NCOMP * DMODEL;

  for (int l = 0; l < NLAYER; ++l) {
    float* qkv[3] = {qb, kb, vb};
    for (int m = 0; m < 3; ++m) {
      size_t idx = (size_t)l * 4 + m;
      custom_neuron_kernel<<<dim3(DMODEL / 16, SEQ / 16), 32, 0, stream>>>(
          x, qkvo_sW + idx * swsz, qkvo_sb + idx * sbsz,
          qkvo_cW + idx * cwsz, qkvo_cb + idx * cbsz,
          qkv[m], DMODEL, DMODEL, 0);
    }
    attn_scores_kernel<<<dim3(SEQ / 16, SEQ / 16, NHEAD), 32, 0, stream>>>(qb, kb, sc);
    softmax_kernel<<<NHEAD * SEQ, 256, 0, stream>>>(sc);
    attn_ctx_kernel<<<dim3(DHEAD / 16, SEQ / 16, NHEAD), 32, 0, stream>>>(sc, vb, cx);
    {
      size_t idx = (size_t)l * 4 + 3;
      custom_neuron_kernel<<<dim3(DMODEL / 16, SEQ / 16), 32, 0, stream>>>(
          cx, qkvo_sW + idx * swsz, qkvo_sb + idx * sbsz,
          qkvo_cW + idx * cwsz, qkvo_cb + idx * cbsz,
          fb, DMODEL, DMODEL, 0);
    }
    add_ln_kernel<<<SEQ, 256, 0, stream>>>(x, fb, ln1_g + (size_t)l * DMODEL,
                                           ln1_b + (size_t)l * DMODEL);

    custom_neuron_kernel<<<dim3(FFDIM / 16, SEQ / 16), 32, 0, stream>>>(
        x,
        l1_sW + (size_t)l * FFDIM * NCOMP * DMODEL,
        l1_sb + (size_t)l * FFDIM * NCOMP,
        l1_cW + (size_t)l * NCOMP * FFDIM * DMODEL,
        l1_cb + (size_t)l * NCOMP * FFDIM,
        hb, FFDIM, DMODEL, 1);

    custom_neuron_kernel<<<dim3(DMODEL / 16, SEQ / 16), 32, 0, stream>>>(
        hb,
        l2_sW + (size_t)l * DMODEL * NCOMP * FFDIM,
        l2_sb + (size_t)l * DMODEL * NCOMP,
        l2_cW + (size_t)l * NCOMP * DMODEL * FFDIM,
        l2_cb + (size_t)l * NCOMP * DMODEL,
        fb, DMODEL, FFDIM, 0);

    add_ln_kernel<<<SEQ, 256, 0, stream>>>(x, fb, ln2_g + (size_t)l * DMODEL,
                                           ln2_b + (size_t)l * DMODEL);
  }

  // Final vocab projection straight into d_out: [S, V] (B=1)
  custom_neuron_kernel<<<dim3(VOCAB / 16, SEQ / 16), 32, 0, stream>>>(
      x, out_sW, out_sb, out_cW, out_cb, (float*)d_out, VOCAB, DMODEL, 0);
}
